// GCN_73280732004500
// MI455X (gfx1250) — matrix operand order repaired and checked by reading, not compiled, not run on hardware
//
#include <hip/hip_runtime.h>
#include <hip/hip_bf16.h>

typedef __attribute__((ext_vector_type(16))) __bf16 v16bf;
typedef __attribute__((ext_vector_type(8)))  float  v8f;

// ---------------------------------------------------------------------------
// Normalization pre-pass: deg[n] = 1 (self loop) + sum_{e: dst=n} ew[e];
// dis[n] = rsqrt(deg). Computed once, reused by all 3 GCN layers.
// ---------------------------------------------------------------------------
__global__ void k_init(float* __restrict__ deg, float* __restrict__ sums,
                       float* __restrict__ cnts, int nNodes) {
  int i = blockIdx.x * blockDim.x + threadIdx.x;
  if (i < nNodes) deg[i] = 1.0f;      // self-loop weight
  if (i < 64 * 64) sums[i] = 0.0f;    // pooling accumulators (re-zeroed per launch)
  if (i < 64) cnts[i] = 0.0f;
}

__global__ void k_deg(const int* __restrict__ dst, const float* __restrict__ ew,
                      float* __restrict__ deg, int nE) {
  int e = blockIdx.x * blockDim.x + threadIdx.x;
  if (e < nE) atomicAdd(&deg[dst[e]], ew[e]);
}

__global__ void k_dis(const float* __restrict__ deg, float* __restrict__ dis, int n) {
  int i = blockIdx.x * blockDim.x + threadIdx.x;
  if (i < n) {
    float d = deg[i];
    dis[i] = (d > 0.0f) ? rsqrtf(d) : 0.0f;
  }
}

// ---------------------------------------------------------------------------
// Pack W[KD x ND] (f32, row major) into WMMA-B-fragment-exact bf16 hi/lo:
// Wp[kt][n][k] , k = 0..31 contiguous  ->  a B fragment for (n, kh) is one
// aligned 32-byte load at ((kt*ND + n)*32 + 16*kh).
// hi = truncated top 16 bits (exact split), lo = RTNE bf16 of the residual.
// ---------------------------------------------------------------------------
template <int KD, int ND>
__global__ void k_packW(const float* __restrict__ W,
                        unsigned short* __restrict__ WpH,
                        unsigned short* __restrict__ WpL) {
  constexpr int NKT = KD / 32;
  int i = blockIdx.x * blockDim.x + threadIdx.x;
  if (i >= NKT * ND) return;
  int kt = i / ND, n = i % ND;
  unsigned short* oh = WpH + (size_t)i * 32;
  unsigned short* ol = WpL + (size_t)i * 32;
#pragma unroll 8
  for (int k = 0; k < 32; ++k) {
    float f = W[(size_t)(kt * 32 + k) * ND + n];
    unsigned int u = __float_as_uint(f);
    float hf = __uint_as_float(u & 0xFFFF0000u);
    __bf16 lb = (__bf16)(f - hf);
    oh[k] = (unsigned short)(u >> 16);
    ol[k] = __builtin_bit_cast(unsigned short, lb);
  }
}

// ---------------------------------------------------------------------------
// LDS-free split-bf16 WMMA GEMM: Out[M x ND] = X[M x KD] @ W[KD x ND].
// acc += Ahi*Bhi + Alo*Bhi + Ahi*Blo (f32 accumulate) -> ~fp32 accuracy.
// Block = 256 threads = 8 independent waves; wave owns 16 rows x ND.
// A fragment: one contiguous 64B f32 run per lane (k = 16kh+s), hi/lo split in registers.
// B fragment: one aligned 32B v16bf load from pre-packed Wp (WGP$-resident).
// ---------------------------------------------------------------------------
template <int KD, int ND>
__global__ __launch_bounds__(256) void k_gemm(
    const float* __restrict__ X, const unsigned short* __restrict__ WpH,
    const unsigned short* __restrict__ WpL, float* __restrict__ Out, int M) {
  constexpr int NT = ND / 16;
  const int tid  = threadIdx.x;
  const int wave = tid >> 5;
  const int lane = tid & 31;
  const int mloc = lane & 15;  // A: row-in-tile. B/C: column-in-tile.
  const int kh   = lane >> 4;  // half-select per ISA layout tables

  int row = blockIdx.x * 128 + wave * 16 + mloc;
  if (row >= M) row = M - 1;                 // clamp loads; stores are guarded
  const float* __restrict__ xrow = X + (size_t)row * KD;

  v8f acc[NT];
#pragma unroll
  for (int t = 0; t < NT; ++t)
#pragma unroll
    for (int r = 0; r < 8; ++r) acc[t][r] = 0.0f;

  for (int kt = 0; kt < KD / 32; ++kt) {
    const int k0 = kt * 32;
    // ---- A fragment: K = [16kh..16kh+15] of this row ----
    const float4 a0 = *reinterpret_cast<const float4*>(xrow + k0 + 16 * kh);
    const float4 a1 = *reinterpret_cast<const float4*>(xrow + k0 + 16 * kh + 4);
    const float4 a2 = *reinterpret_cast<const float4*>(xrow + k0 + 16 * kh + 8);
    const float4 a3 = *reinterpret_cast<const float4*>(xrow + k0 + 16 * kh + 12);
    float f[16] = {a0.x, a0.y, a0.z, a0.w, a1.x, a1.y, a1.z, a1.w,
                   a2.x, a2.y, a2.z, a2.w, a3.x, a3.y, a3.z, a3.w};
    union { v16bf bf; unsigned int u[8]; } ah, al;
#pragma unroll
    for (int j = 0; j < 8; ++j) {
      unsigned int u0 = __float_as_uint(f[2 * j]);
      unsigned int u1 = __float_as_uint(f[2 * j + 1]);
      ah.u[j] = (u0 >> 16) | (u1 & 0xFFFF0000u);           // exact truncation split
      float h0 = __uint_as_float(u0 & 0xFFFF0000u);
      float h1 = __uint_as_float(u1 & 0xFFFF0000u);
      __bf16 l0 = (__bf16)(f[2 * j] - h0);                 // RTNE residual
      __bf16 l1 = (__bf16)(f[2 * j + 1] - h1);
      al.u[j] = (unsigned int)__builtin_bit_cast(unsigned short, l0) |
                ((unsigned int)__builtin_bit_cast(unsigned short, l1) << 16);
    }
    // ---- B fragments (pre-packed) + 3 WMMAs per N-tile ----
#pragma unroll
    for (int t = 0; t < NT; ++t) {
      const size_t boff = ((size_t)(kt * ND + t * 16 + mloc)) * 32 + 16 * kh;
      v16bf bh = *reinterpret_cast<const v16bf*>(WpH + boff);
      v16bf bl = *reinterpret_cast<const v16bf*>(WpL + boff);
      acc[t] = __builtin_amdgcn_wmma_f32_16x16x32_bf16(
          false, ah.bf, false, bh, (short)0, acc[t], false, false);
      acc[t] = __builtin_amdgcn_wmma_f32_16x16x32_bf16(
          false, al.bf, false, bh, (short)0, acc[t], false, false);
      acc[t] = __builtin_amdgcn_wmma_f32_16x16x32_bf16(
          false, ah.bf, false, bl, (short)0, acc[t], false, false);
    }
  }

  // ---- store D per ISA C/D layout: lane = (n=mloc, kh), vgpr r = m & 7 ----
  const int mBase = blockIdx.x * 128 + wave * 16 + kh * 8;
#pragma unroll
  for (int r = 0; r < 8; ++r) {
    int grow = mBase + r;
    if (grow < M) {
#pragma unroll
      for (int t = 0; t < NT; ++t)
        Out[(size_t)grow * ND + t * 16 + mloc] = acc[t][r];
    }
  }
}

// ---------------------------------------------------------------------------
// Self-loop term: out[n] = h[n] * dis[n]^2  (also serves as output init)
// ---------------------------------------------------------------------------
template <int C>
__global__ void k_selfloop(const float* __restrict__ h, const float* __restrict__ dis,
                           float* __restrict__ outb, int M) {
  int i = blockIdx.x * blockDim.x + threadIdx.x;
  if (i < M * (C / 4)) {
    int row = i / (C / 4);
    float s = dis[row]; s *= s;
    float4 v = reinterpret_cast<const float4*>(h)[i];
    reinterpret_cast<float4*>(outb)[i] =
        make_float4(v.x * s, v.y * s, v.z * s, v.w * s);
  }
}

// ---------------------------------------------------------------------------
// Edge scatter: out[dst] += h[src] * (dis[src]*ew*dis[dst]).  One wave32 per
// edge, C/32 channels per lane, f32 atomics resolving in L2.
// ---------------------------------------------------------------------------
template <int C>
__global__ void k_scatter(const float* __restrict__ h, const int* __restrict__ src,
                          const int* __restrict__ dst, const float* __restrict__ ew,
                          const float* __restrict__ dis, float* __restrict__ outb,
                          int nE) {
  constexpr int PER = C / 32;
  int lane   = threadIdx.x & 31;
  int wid    = (blockIdx.x * blockDim.x + threadIdx.x) >> 5;
  int stride = (gridDim.x * blockDim.x) >> 5;
  for (int e = wid; e < nE; e += stride) {
    int s = src[e], d = dst[e];
    float nrm = dis[s] * ew[e] * dis[d];
    const float* hp = h + (size_t)s * C + lane * PER;
    float*       op = outb + (size_t)d * C + lane * PER;
#pragma unroll
    for (int j = 0; j < PER; ++j) atomicAdd(&op[j], hp[j] * nrm);
  }
}

template <int C, bool RELU>
__global__ void k_finalize(float* __restrict__ outb, const float* __restrict__ bias,
                           int M) {
  int i = blockIdx.x * blockDim.x + threadIdx.x;
  if (i < M * C) {
    float v = outb[i] + bias[i & (C - 1)];
    if (RELU) v = fmaxf(v, 0.0f);
    outb[i] = v;
  }
}

// ---------------------------------------------------------------------------
// Global mean pool (64 graphs x 64 ch) + 64->3 linear head
// ---------------------------------------------------------------------------
__global__ void k_pool(const float* __restrict__ h, const int* __restrict__ batch,
                       float* __restrict__ sums, float* __restrict__ cnts, int M) {
  int i = blockIdx.x * blockDim.x + threadIdx.x;
  if (i < M * 64) {
    int n = i >> 6, c = i & 63;
    int g = batch[n];
    atomicAdd(&sums[g * 64 + c], h[i]);
    if (c == 0) atomicAdd(&cnts[g], 1.0f);
  }
}

__global__ void k_head(const float* __restrict__ sums, const float* __restrict__ cnts,
                       const float* __restrict__ Wl, const float* __restrict__ bl,
                       float* __restrict__ out) {
  int g = threadIdx.x;
  if (g < 64) {
    float inv = 1.0f / fmaxf(cnts[g], 1.0f);
    float o0 = bl[0], o1 = bl[1], o2 = bl[2];
    for (int c = 0; c < 64; ++c) {
      float p = sums[g * 64 + c] * inv;
      o0 += p * Wl[c * 3 + 0];
      o1 += p * Wl[c * 3 + 1];
      o2 += p * Wl[c * 3 + 2];
    }
    out[g * 3 + 0] = o0; out[g * 3 + 1] = o1; out[g * 3 + 2] = o2;
  }
}

// ---------------------------------------------------------------------------
extern "C" void kernel_launch(void* const* d_in, const int* in_sizes, int n_in,
                              void* d_out, int out_size, void* d_ws, size_t ws_size,
                              hipStream_t stream) {
  (void)n_in; (void)out_size; (void)ws_size;
  const float* x     = (const float*)d_in[0];
  const int*   ei    = (const int*)d_in[1];
  const float* ew    = (const float*)d_in[2];
  const int*   batch = (const int*)d_in[3];
  const float* W1 = (const float*)d_in[4];
  const float* b1 = (const float*)d_in[5];
  const float* W2 = (const float*)d_in[6];
  const float* b2 = (const float*)d_in[7];
  const float* W3 = (const float*)d_in[8];
  const float* b3 = (const float*)d_in[9];
  const float* Wl = (const float*)d_in[10];
  const float* bl = (const float*)d_in[11];
  float* out = (float*)d_out;

  const int M  = in_sizes[3];   // nodes (batch vector length)
  const int nE = in_sizes[2];   // edges (edge_weight length)
  const int* src = ei;
  const int* dst = ei + nE;

  // workspace layout (128B-aligned regions)
  auto al = [](size_t bytes) { return (bytes + 127) & ~size_t(127); };
  char* ws = (char*)d_ws;
  size_t o = 0;
  float* deg  = (float*)(ws + o); o += al((size_t)M * 4);
  float* dis  = (float*)(ws + o); o += al((size_t)M * 4);
  float* T    = (float*)(ws + o); o += al((size_t)M * 128 * 4);  // GEMM out
  float* A    = (float*)(ws + o); o += al((size_t)M * 128 * 4);  // H1 / H3
  float* B    = (float*)(ws + o); o += al((size_t)M * 128 * 4);  // H2
  float* sums = (float*)(ws + o); o += al(64 * 64 * 4);
  float* cnts = (float*)(ws + o); o += al(64 * 4);
  // packed weights (bf16 hi/lo, fragment-exact layout)
  unsigned short* Wp1H = (unsigned short*)(ws + o); o += al((size_t)(1056 / 32) * 128 * 32 * 2);
  unsigned short* Wp1L = (unsigned short*)(ws + o); o += al((size_t)(1056 / 32) * 128 * 32 * 2);
  unsigned short* Wp2H = (unsigned short*)(ws + o); o += al((size_t)(128 / 32) * 128 * 32 * 2);
  unsigned short* Wp2L = (unsigned short*)(ws + o); o += al((size_t)(128 / 32) * 128 * 32 * 2);
  unsigned short* Wp3H = (unsigned short*)(ws + o); o += al((size_t)(128 / 32) * 64 * 32 * 2);
  unsigned short* Wp3L = (unsigned short*)(ws + o); o += al((size_t)(128 / 32) * 64 * 32 * 2);

  const int TB = 256;
  // ---- weight packing (tiny; L2-resident matrices) ----
  k_packW<1056, 128><<<((1056 / 32) * 128 + TB - 1) / TB, TB, 0, stream>>>(W1, Wp1H, Wp1L);
  k_packW<128, 128><<<((128 / 32) * 128 + TB - 1) / TB, TB, 0, stream>>>(W2, Wp2H, Wp2L);
  k_packW<128, 64><<<((128 / 32) * 64 + TB - 1) / TB, TB, 0, stream>>>(W3, Wp3H, Wp3L);

  // ---- normalization (shared by all layers) ----
  k_init<<<(M + TB - 1) / TB, TB, 0, stream>>>(deg, sums, cnts, M);
  k_deg<<<(nE + TB - 1) / TB, TB, 0, stream>>>(dst, ew, deg, nE);
  k_dis<<<(M + TB - 1) / TB, TB, 0, stream>>>(deg, dis, M);

  const int gBlk = (M + 127) / 128;
  // ---- layer 1: 1056 -> 128, relu ----
  k_gemm<1056, 128><<<gBlk, TB, 0, stream>>>(x, Wp1H, Wp1L, T, M);
  k_selfloop<128><<<(M * 32 + TB - 1) / TB, TB, 0, stream>>>(T, dis, A, M);
  k_scatter<128><<<2048, TB, 0, stream>>>(T, src, dst, ew, dis, A, nE);
  k_finalize<128, true><<<(M * 128 + TB - 1) / TB, TB, 0, stream>>>(A, b1, M);
  // ---- layer 2: 128 -> 128, relu ----
  k_gemm<128, 128><<<gBlk, TB, 0, stream>>>(A, Wp2H, Wp2L, T, M);
  k_selfloop<128><<<(M * 32 + TB - 1) / TB, TB, 0, stream>>>(T, dis, B, M);
  k_scatter<128><<<2048, TB, 0, stream>>>(T, src, dst, ew, dis, B, nE);
  k_finalize<128, true><<<(M * 128 + TB - 1) / TB, TB, 0, stream>>>(B, b2, M);
  // ---- layer 3: 128 -> 64, no relu ----
  k_gemm<128, 64><<<gBlk, TB, 0, stream>>>(B, Wp3H, Wp3L, T, M);
  k_selfloop<64><<<(M * 16 + TB - 1) / TB, TB, 0, stream>>>(T, dis, A, M);
  k_scatter<64><<<2048, TB, 0, stream>>>(T, src, dst, ew, dis, A, nE);
  k_finalize<64, false><<<(M * 64 + TB - 1) / TB, TB, 0, stream>>>(A, b3, M);
  // ---- mean pool + linear head ----
  k_pool<<<(M * 64 + TB - 1) / TB, TB, 0, stream>>>(A, batch, sums, cnts, M);
  k_head<<<1, 64, 0, stream>>>(sums, cnts, Wl, bl, out);
}
